// MixtureOfExperts_26534307955388
// MI455X (gfx1250) — compile-verified, matrix operand order checked by reading, not run on hardware
//
#include <hip/hip_runtime.h>
#include <hip/hip_bf16.h>

// ---------------------------------------------------------------------------
// MoE (top-4 of 16 routed experts + shared expert, SwiGLU) for MI455X/gfx1250.
// bf16 WMMA (v_wmma_f32_16x16x32_bf16) with fp32 accumulation.
// Round 2: register-blocked wave tiles (A: 32x32 dual-matrix, B: 32x64) to
// raise L2 arithmetic intensity from ~11 to ~21 FLOP/byte.
// ---------------------------------------------------------------------------

#define H_DIM 1024
#define I_DIM 512
#define E_NUM 16
#define T_NUM 2048
#define EP1   17            // 16 routed experts + 1 shared
#define KTOT  (EP1 * I_DIM) // 8704

typedef __attribute__((ext_vector_type(8)))  __bf16 v8bf;
typedef __attribute__((ext_vector_type(16))) __bf16 v16bf;
typedef __attribute__((ext_vector_type(8)))  float  v8f;

union Frag { v16bf v; v8bf h[2]; };

// Load one WMMA operand fragment (16 bf16 / lane = 2 x 16B loads).
// base = rowptr + k0 + khalf; K layout: {0..7,16..23} for lanes 0-15,
// {8..15,24..31} for lanes 16-31 (khalf encodes the +8).
__device__ __forceinline__ v16bf load_frag(const __bf16* base) {
    Frag f;
    f.h[0] = *(const v8bf*)(base);
    f.h[1] = *(const v8bf*)(base + 16);
    return f.v;
}

#define WMMA_BF16(A, B, C) \
    __builtin_amdgcn_wmma_f32_16x16x32_bf16(false, (A), false, (B), (short)0, (C), false, false)

// ---------------------------------------------------------------------------
// Prep: fp32 -> bf16 straight copy (activations, row-major [T,H])
// ---------------------------------------------------------------------------
__global__ void cvt_copy_kernel(const float* __restrict__ src,
                                __bf16* __restrict__ dst, long long n) {
    long long stride = (long long)gridDim.x * blockDim.x;
    for (long long i = (long long)blockIdx.x * blockDim.x + threadIdx.x; i < n; i += stride)
        dst[i] = (__bf16)src[i];
}

// ---------------------------------------------------------------------------
// Prep: fp32 [batch][R][C] -> bf16 [batch][C][R]  (gives K-contiguous B^T)
// ---------------------------------------------------------------------------
__global__ void cvt_transpose_kernel(const float* __restrict__ src,
                                     __bf16* __restrict__ dst,
                                     int R, int C, int batch) {
    long long total  = (long long)batch * R * C;
    long long stride = (long long)gridDim.x * blockDim.x;
    for (long long i = (long long)blockIdx.x * blockDim.x + threadIdx.x; i < total; i += stride) {
        long long rc  = (long long)R * C;
        long long b   = i / rc;
        long long rem = i - b * rc;
        long long c   = rem / R;
        long long r   = rem - c * R;
        dst[i] = (__bf16)src[(b * R + r) * C + c];
    }
}

// ---------------------------------------------------------------------------
// Routing: sigmoid(x @ gate_w^T), top-4, normalized -> dense comb[T][17]
// (slot 16 = shared expert, weight 1.0). One 64-thread block per token.
// ---------------------------------------------------------------------------
__global__ void route_kernel(const float* __restrict__ x,
                             const float* __restrict__ gate_w,
                             float* __restrict__ comb) {
    const int t   = blockIdx.x;
    const int tid = threadIdx.x;
    const int e   = tid & 15;        // expert
    const int sl  = tid >> 4;        // 4 slices of 256 over H

    const float* xr = x + (long long)t * H_DIM;
    const float* gr = gate_w + (long long)e * H_DIM;
    float p = 0.f;
    for (int h = sl * 256; h < sl * 256 + 256; ++h) p += xr[h] * gr[h];

    __shared__ float part[64];
    __shared__ float sc[16];
    part[tid] = p;
    __syncthreads();
    if (tid < 16) {
        float s = part[tid] + part[tid + 16] + part[tid + 32] + part[tid + 48];
        sc[tid] = 1.f / (1.f + __expf(-s));
    }
    __syncthreads();
    if (tid == 0) {
        bool used[16];
        for (int i = 0; i < 16; ++i) used[i] = false;
        int   id[4];
        float w[4];
        float sum = 0.f;
        for (int k = 0; k < 4; ++k) {
            int best = 0; float bv = -1e30f;
            for (int ee = 0; ee < 16; ++ee)
                if (!used[ee] && sc[ee] > bv) { bv = sc[ee]; best = ee; }
            used[best] = true; id[k] = best; w[k] = bv; sum += bv;
        }
        float* c = comb + (long long)t * EP1;
        for (int ee = 0; ee < EP1; ++ee) c[ee] = 0.f;
        float inv = 1.f / sum;
        for (int k = 0; k < 4; ++k) c[id[k]] = w[k] * inv;
        c[16] = 1.0f;  // shared expert always on
    }
}

// ---------------------------------------------------------------------------
// Stage A: per "expert" e in [0,16]: g = x@Wg_e, u = x@Wu_e (WMMA bf16),
// h = comb[t,e] * silu(g) * u  -> hcomb[t][e][i] (bf16).
// Wave tile: M=32, N=32, both g and u (8 accumulators, 8 WMMAs / k-step).
// grid = (I/128, T/32, 17), block = 128 (4 waves -> N coverage 128).
// ---------------------------------------------------------------------------
__global__ void moe_up_kernel(const __bf16* __restrict__ xbf,
                              const __bf16* __restrict__ WgT,   // [E][I][H]
                              const __bf16* __restrict__ WuT,   // [E][I][H]
                              const __bf16* __restrict__ sgT,   // [I][H]
                              const __bf16* __restrict__ suT,   // [I][H]
                              const float*  __restrict__ comb,  // [T][17]
                              __bf16* __restrict__ hcomb)       // [T][17][I]
{
    const int lane  = threadIdx.x & 31;
    const int wave  = threadIdx.x >> 5;
    const int e     = blockIdx.z;                       // 0..16
    const int Nbase = blockIdx.x * 128 + wave * 32;     // column in I
    const int Mbase = blockIdx.y * 32;                  // token row

    const int l15   = lane & 15;
    const int khalf = (lane >> 4) << 3;                 // 0 or 8

    const __bf16* Bg = (e < 16) ? (WgT + (long long)e * I_DIM * H_DIM) : sgT;
    const __bf16* Bu = (e < 16) ? (WuT + (long long)e * I_DIM * H_DIM) : suT;

    const __bf16* aRow0  = xbf + (long long)(Mbase + l15)      * H_DIM + khalf;
    const __bf16* aRow1  = xbf + (long long)(Mbase + 16 + l15) * H_DIM + khalf;
    const __bf16* bgRow0 = Bg  + (long long)(Nbase + l15)      * H_DIM + khalf;
    const __bf16* bgRow1 = Bg  + (long long)(Nbase + 16 + l15) * H_DIM + khalf;
    const __bf16* buRow0 = Bu  + (long long)(Nbase + l15)      * H_DIM + khalf;
    const __bf16* buRow1 = Bu  + (long long)(Nbase + 16 + l15) * H_DIM + khalf;

    v8f accG[2][2] = {};
    v8f accU[2][2] = {};
    for (int k0 = 0; k0 < H_DIM; k0 += 32) {
        __builtin_prefetch((const void*)(bgRow0 + k0 + 256), 0, 0);
        v16bf a0  = load_frag(aRow0  + k0);
        v16bf a1  = load_frag(aRow1  + k0);
        v16bf bg0 = load_frag(bgRow0 + k0);
        v16bf bg1 = load_frag(bgRow1 + k0);
        v16bf bu0 = load_frag(buRow0 + k0);
        v16bf bu1 = load_frag(buRow1 + k0);
        accG[0][0] = WMMA_BF16(a0, bg0, accG[0][0]);
        accG[0][1] = WMMA_BF16(a0, bg1, accG[0][1]);
        accG[1][0] = WMMA_BF16(a1, bg0, accG[1][0]);
        accG[1][1] = WMMA_BF16(a1, bg1, accG[1][1]);
        accU[0][0] = WMMA_BF16(a0, bu0, accU[0][0]);
        accU[0][1] = WMMA_BF16(a0, bu1, accU[0][1]);
        accU[1][0] = WMMA_BF16(a1, bu0, accU[1][0]);
        accU[1][1] = WMMA_BF16(a1, bu1, accU[1][1]);
    }

    // C/D layout: element r of lane l -> M = r + (l>=16 ? 8 : 0), N = l&15.
    const int madd = (lane >> 4) << 3;
#pragma unroll
    for (int mi = 0; mi < 2; ++mi) {
#pragma unroll
        for (int r = 0; r < 8; ++r) {
            const int   t = Mbase + mi * 16 + madd + r;
            const float w = comb[(long long)t * EP1 + e];
            __bf16* hrow = hcomb + ((long long)t * EP1 + e) * I_DIM;
#pragma unroll
            for (int ni = 0; ni < 2; ++ni) {
                float g  = accG[mi][ni][r];
                float u  = accU[mi][ni][r];
                float hv = w * u * (g / (1.f + __expf(-g)));  // comb folded in
                hrow[Nbase + ni * 16 + l15] = (__bf16)hv;
            }
        }
    }
}

// ---------------------------------------------------------------------------
// Stage B: out[t,n] = sum_e  hcomb[t,e,:] @ Wd_e[:,n]  (+ shared via sdT).
// Wave tile: M=32, N=64 (8 accumulators, 8 WMMAs / k-step).
// K dimension = 17*512 = 8704. grid = (H/256, T/32), block = 128.
// ---------------------------------------------------------------------------
__global__ void moe_down_kernel(const __bf16* __restrict__ hcomb, // [T][17][I]
                                const __bf16* __restrict__ WdT,   // [E][H][I]
                                const __bf16* __restrict__ sdT,   // [H][I]
                                float* __restrict__ out)          // [T][H]
{
    const int lane  = threadIdx.x & 31;
    const int wave  = threadIdx.x >> 5;
    const int Nbase = blockIdx.x * 256 + wave * 64;     // column in H
    const int Mbase = blockIdx.y * 32;                  // token row

    const int l15   = lane & 15;
    const int khalf = (lane >> 4) << 3;

    const __bf16* aRow0 = hcomb + (long long)(Mbase + l15)      * KTOT + khalf;
    const __bf16* aRow1 = hcomb + (long long)(Mbase + 16 + l15) * KTOT + khalf;

    v8f acc[2][4] = {};
    for (int e = 0; e < EP1; ++e) {
        const __bf16* bRow[4];
#pragma unroll
        for (int ni = 0; ni < 4; ++ni) {
            const int col = Nbase + ni * 16 + l15;
            bRow[ni] = (e < 16)
                ? (WdT + ((long long)e * H_DIM + col) * I_DIM + khalf)
                : (sdT + (long long)col * I_DIM + khalf);
        }
        const __bf16* aE0 = aRow0 + (long long)e * I_DIM;
        const __bf16* aE1 = aRow1 + (long long)e * I_DIM;
        for (int kin = 0; kin < I_DIM; kin += 32) {
            __builtin_prefetch((const void*)(bRow[0] + kin + 256), 0, 0);
            v16bf a0 = load_frag(aE0 + kin);
            v16bf a1 = load_frag(aE1 + kin);
#pragma unroll
            for (int ni = 0; ni < 4; ++ni) {
                v16bf b = load_frag(bRow[ni] + kin);
                acc[0][ni] = WMMA_BF16(a0, b, acc[0][ni]);
                acc[1][ni] = WMMA_BF16(a1, b, acc[1][ni]);
            }
        }
    }

    const int madd = (lane >> 4) << 3;
#pragma unroll
    for (int mi = 0; mi < 2; ++mi) {
#pragma unroll
        for (int r = 0; r < 8; ++r) {
            const int t = Mbase + mi * 16 + madd + r;
            float* orow = out + (long long)t * H_DIM;
#pragma unroll
            for (int ni = 0; ni < 4; ++ni)
                orow[Nbase + ni * 16 + l15] = acc[mi][ni][r];
        }
    }
}

// ---------------------------------------------------------------------------
// Host-side launch. Workspace layout (bytes, 256-aligned by construction):
//   xbf   4 MiB | WgT 16 MiB | WuT 16 MiB | WdT 16 MiB |
//   sgT/suT/sdT 1 MiB each | comb 136 KiB | hcomb 34 MiB   (~89 MiB total)
// ---------------------------------------------------------------------------
extern "C" void kernel_launch(void* const* d_in, const int* in_sizes, int n_in,
                              void* d_out, int out_size, void* d_ws, size_t ws_size,
                              hipStream_t stream) {
    const float* x    = (const float*)d_in[0];  // [T,H]
    const float* gw   = (const float*)d_in[1];  // [E,H]
    const float* Wg   = (const float*)d_in[2];  // [E,H,I]
    const float* Wu   = (const float*)d_in[3];  // [E,H,I]
    const float* Wd   = (const float*)d_in[4];  // [E,I,H]
    const float* sg   = (const float*)d_in[5];  // [H,I]
    const float* su   = (const float*)d_in[6];  // [H,I]
    const float* sd   = (const float*)d_in[7];  // [I,H]
    float* out = (float*)d_out;

    char* ws = (char*)d_ws;
    size_t off = 0;
    __bf16* xbf = (__bf16*)(ws + off); off += (size_t)T_NUM * H_DIM * 2;            // 4 MiB
    __bf16* WgT = (__bf16*)(ws + off); off += (size_t)E_NUM * H_DIM * I_DIM * 2;    // 16 MiB
    __bf16* WuT = (__bf16*)(ws + off); off += (size_t)E_NUM * H_DIM * I_DIM * 2;    // 16 MiB
    __bf16* WdT = (__bf16*)(ws + off); off += (size_t)E_NUM * H_DIM * I_DIM * 2;    // 16 MiB
    __bf16* sgT = (__bf16*)(ws + off); off += (size_t)H_DIM * I_DIM * 2;            // 1 MiB
    __bf16* suT = (__bf16*)(ws + off); off += (size_t)H_DIM * I_DIM * 2;            // 1 MiB
    __bf16* sdT = (__bf16*)(ws + off); off += (size_t)H_DIM * I_DIM * 2;            // 1 MiB
    float*  comb = (float*)(ws + off); off += (size_t)T_NUM * EP1 * 4;              // 136 KiB
    __bf16* hcomb = (__bf16*)(ws + off);                                            // 34 MiB

    // --- prep: bf16 conversion (+ transpose to [N][K] for all B operands) ---
    cvt_copy_kernel<<<1024, 256, 0, stream>>>(x, xbf, (long long)T_NUM * H_DIM);
    cvt_transpose_kernel<<<2048, 256, 0, stream>>>(Wg, WgT, H_DIM, I_DIM, E_NUM);
    cvt_transpose_kernel<<<2048, 256, 0, stream>>>(Wu, WuT, H_DIM, I_DIM, E_NUM);
    cvt_transpose_kernel<<<2048, 256, 0, stream>>>(Wd, WdT, I_DIM, H_DIM, E_NUM);
    cvt_transpose_kernel<<<512, 256, 0, stream>>>(sg, sgT, H_DIM, I_DIM, 1);
    cvt_transpose_kernel<<<512, 256, 0, stream>>>(su, suT, H_DIM, I_DIM, 1);
    cvt_transpose_kernel<<<512, 256, 0, stream>>>(sd, sdT, I_DIM, H_DIM, 1);

    // --- routing ---
    route_kernel<<<T_NUM, 64, 0, stream>>>(x, gw, comb);

    // --- stage A: gate/up GEMMs + SwiGLU + combine-weight fold ---
    {
        dim3 grid(I_DIM / 128, T_NUM / 32, EP1);
        moe_up_kernel<<<grid, 128, 0, stream>>>(xbf, WgT, WuT, sgT, suT, comb, hcomb);
    }

    // --- stage B: down GEMM over K = 17*512, writes every out element ---
    {
        dim3 grid(H_DIM / 256, T_NUM / 32);
        moe_down_kernel<<<grid, 128, 0, stream>>>(hcomb, WdT, sdT, out);
    }
}
